// SimpleSubModule_52742198395363
// MI455X (gfx1250) — 8-bit matrix operand order repaired and checked by reading, compiled, not run on hardware
//
#include <hip/hip_runtime.h>
#include <hip/hip_bf16.h>

typedef __attribute__((ext_vector_type(8))) int v8i;

union FragI { v8i v; int i[8]; };

__global__ void init_ws_kernel(unsigned int* __restrict__ umax) {
    if (threadIdx.x < 2) umax[threadIdx.x] = 0u;
}

__global__ void absmax_kernel(const float* __restrict__ x, size_t n4,
                              unsigned int* __restrict__ out) {
    size_t tid    = (size_t)blockIdx.x * blockDim.x + threadIdx.x;
    size_t stride = (size_t)gridDim.x * blockDim.x;
    float m = 0.0f;
    for (size_t i = tid; i < n4; i += stride) {
        float4 v = ((const float4*)x)[i];
        m = fmaxf(m, fmaxf(fmaxf(fabsf(v.x), fabsf(v.y)),
                           fmaxf(fabsf(v.z), fabsf(v.w))));
    }
    #pragma unroll
    for (int off = 16; off > 0; off >>= 1)
        m = fmaxf(m, __shfl_xor(m, off, 32));
    if ((threadIdx.x & 31) == 0)
        atomicMax(out, __float_as_uint(m));
}

__global__ void quant_kernel(const float* __restrict__ x,
                             signed char* __restrict__ q, size_t n4,
                             const unsigned int* __restrict__ umax,
                             float div, float qmin, float qmax) {
    float s   = __uint_as_float(*umax) / div;
    float inv = (s > 0.0f) ? (1.0f / s) : 0.0f;
    size_t tid    = (size_t)blockIdx.x * blockDim.x + threadIdx.x;
    size_t stride = (size_t)gridDim.x * blockDim.x;
    for (size_t i = tid; i < n4; i += stride) {
        float4 v = ((const float4*)x)[i];
        int a = (int)fminf(fmaxf(rintf(v.x * inv), qmin), qmax);
        int b = (int)fminf(fmaxf(rintf(v.y * inv), qmin), qmax);
        int c = (int)fminf(fmaxf(rintf(v.z * inv), qmin), qmax);
        int d = (int)fminf(fmaxf(rintf(v.w * inv), qmin), qmax);
        ((int*)q)[i] = (a & 255) | ((b & 255) << 8) | ((c & 255) << 16) | (d << 24);
    }
}

#define BM 128
#define BN 128
#define KSTEP 64

__global__ __launch_bounds__(256)
void gemm_iu8_kernel(const signed char* __restrict__ qx,
                     const signed char* __restrict__ qw,
                     const float* __restrict__ bias,
                     const float* __restrict__ mul_val,
                     const unsigned int* __restrict__ umax,
                     float* __restrict__ out,
                     int M, int N, int K) {
    const int lane     = threadIdx.x & 31;
    const int wave     = threadIdx.x >> 5;
    const int waveM    = wave >> 1;
    const int waveN    = wave & 1;
    const int laneHalf = lane >> 4;
    const int lane16   = lane & 15;

    const int waveRow = blockIdx.y * BM + waveM * 32;
    const int waveCol = blockIdx.x * BN + waveN * 64;

    const v8i vzero = {};
    v8i acc[2][4];
    #pragma unroll
    for (int mi = 0; mi < 2; ++mi)
        #pragma unroll
        for (int ni = 0; ni < 4; ++ni)
            acc[mi][ni] = vzero;

    const signed char* aptr[2];
    aptr[0] = qx + (size_t)(waveRow + lane16) * K + laneHalf * 16;
    aptr[1] = aptr[0] + (size_t)16 * K;

    const signed char* bptr[4];
    #pragma unroll
    for (int ni = 0; ni < 4; ++ni)
        bptr[ni] = qw + (size_t)(waveCol + ni * 16 + lane16) * K + laneHalf * 16;

    for (int k = 0; k < K; k += KSTEP) {
        FragI a[2];
        #pragma unroll
        for (int mi = 0; mi < 2; ++mi) {
            const signed char* p = aptr[mi] + k;
            #pragma unroll
            for (int j = 0; j < 4; ++j) {
                int2 t = *(const int2*)(p + j * 8 + (j >> 1) * 16);
                a[mi].i[2 * j]     = t.x;
                a[mi].i[2 * j + 1] = t.y;
            }
        }
        FragI b[4];
        #pragma unroll
        for (int ni = 0; ni < 4; ++ni) {
            const signed char* p = bptr[ni] + k;
            int4 t0 = *(const int4*)(p);
            int4 t1 = *(const int4*)(p + 32);
            b[ni].i[0] = t0.x; b[ni].i[1] = t0.y; b[ni].i[2] = t0.z; b[ni].i[3] = t0.w;
            b[ni].i[4] = t1.x; b[ni].i[5] = t1.y; b[ni].i[6] = t1.z; b[ni].i[7] = t1.w;
        }
        #pragma unroll
        for (int mi = 0; mi < 2; ++mi)
            #pragma unroll
            for (int ni = 0; ni < 4; ++ni)
                acc[mi][ni] = __builtin_amdgcn_wmma_i32_16x16x64_iu8(
                     true, a[mi].v,  true, b[ni].v,
                    acc[mi][ni],  false,  false);
    }

    const float sx = __uint_as_float(umax[0]) * (1.0f / 128.0f);
    const float sw = __uint_as_float(umax[1]) * (1.0f / 127.0f);
    const float mv = mul_val[0];
    const float cs = mv * sx * sw;

    float bvals[4];
    #pragma unroll
    for (int ni = 0; ni < 4; ++ni)
        bvals[ni] = mv * bias[waveCol + ni * 16 + lane16];

    #pragma unroll
    for (int mi = 0; mi < 2; ++mi) {
        const int rowBase = waveRow + mi * 16 + laneHalf * 8;
        #pragma unroll
        for (int ni = 0; ni < 4; ++ni) {
            const int col = waveCol + ni * 16 + lane16;
            FragI c; c.v = acc[mi][ni];
            #pragma unroll
            for (int r = 0; r < 8; ++r)
                out[(size_t)(rowBase + r) * N + col] = cs * (float)c.i[r] + bvals[ni];
        }
    }
}

extern "C" void kernel_launch(void* const* d_in, const int* in_sizes, int n_in,
                              void* d_out, int out_size, void* d_ws, size_t ws_size,
                              hipStream_t stream) {
    const float* x    = (const float*)d_in[0];
    const float* w    = (const float*)d_in[1];
    const float* bias = (const float*)d_in[2];
    const float* mulv = (const float*)d_in[3];

    const int OUT = in_sizes[2];
    const int IN  = in_sizes[1] / OUT;
    const int B   = in_sizes[0] / IN;
    const size_t nx = (size_t)B * IN;
    const size_t nw = (size_t)OUT * IN;

    unsigned int* umax = (unsigned int*)d_ws;
    signed char*  qx   = (signed char*)d_ws + 256;
    signed char*  qw   = qx + nx;

    init_ws_kernel<<<1, 32, 0, stream>>>(umax);
    absmax_kernel<<<512, 256, 0, stream>>>(x, nx / 4, &umax[0]);
    absmax_kernel<<<512, 256, 0, stream>>>(w, nw / 4, &umax[1]);
    quant_kernel<<<1024, 256, 0, stream>>>(x, qx, nx / 4, &umax[0], 128.0f, -128.0f, 127.0f);
    quant_kernel<<<1024, 256, 0, stream>>>(w, qw, nw / 4, &umax[1], 127.0f, -127.0f, 127.0f);

    dim3 grid(OUT / BN, B / BM);
    gemm_iu8_kernel<<<grid, 256, 0, stream>>>(qx, qw, bias, mulv, umax,
                                              (float*)d_out, B, OUT, IN);
}
